// SelfCrossAttention_18786186953468
// MI455X (gfx1250) — matrix operand order repaired and checked by reading, not compiled, not run on hardware
//
#include <hip/hip_runtime.h>

typedef _Float16 h16;
typedef __attribute__((ext_vector_type(16))) _Float16 v16h;
typedef __attribute__((ext_vector_type(8)))  float    v8f;
typedef __attribute__((ext_vector_type(4)))  float    f32x4;
typedef __attribute__((ext_vector_type(4)))  unsigned int u32x4;

#define RES_    64
#define DIM_    256
#define HEADS_  8
#define HD_     32
#define L_      4096
#define BATCH_  8
#define SCALE_  0.17677669529663687f   // 32^-0.5

// ---------------------------------------------------------------------------
// Kernel 1: k_sum[bh][m][c] = sum over 8 row-bands of k, output f16 workspace.
// One thread per 4 consecutive channels. k is read exactly once (33.5 MB).
// ---------------------------------------------------------------------------
__global__ __launch_bounds__(256) void sca_ksum_kernel(const float* __restrict__ kptr,
                                                       h16* __restrict__ ws) {
  int idx  = blockIdx.x * 256 + threadIdx.x;   // 64*512*8 = 262144 total
  int cg   = idx & 7;
  int m    = (idx >> 3) & 511;                 // m = r*64 + col
  int bh   = idx >> 12;
  int b    = bh >> 3;
  int head = bh & 7;
  // band 0 spatial == m; band offset = band*512 rows
  const float* base = kptr + ((size_t)b * L_ + m) * DIM_ + head * HD_ + cg * 4;
  f32x4 s = {0.f, 0.f, 0.f, 0.f};
#pragma unroll
  for (int band = 0; band < 8; ++band)
    s += *(const f32x4*)(base + (size_t)band * 512 * DIM_);
  h16* o = ws + ((size_t)bh * 512 + m) * HD_ + cg * 4;
  o[0] = (h16)s[0]; o[1] = (h16)s[1]; o[2] = (h16)s[2]; o[3] = (h16)s[3];
}

// ---------------------------------------------------------------------------
// WMMA helpers (gfx1250 V_WMMA_F32_16X16X32_F16, wave32)
// ---------------------------------------------------------------------------
__device__ __forceinline__ v8f wmma_f16(v16h a, v16h b, v8f c) {
  return __builtin_amdgcn_wmma_f32_16x16x32_f16(false, a, false, b,
                                                (short)0, c, false, false);
}

// B operand (K x 16), column-contiguous storage: colbase[col*rowStride + k].
// lanes 0-15: col=lane, K=kLow..kLow+15 ; lanes 16-31: col=lane-16, K=kLow+16..+31
__device__ __forceinline__ v16h load_B_lds(const h16* colbase, int rowStride,
                                           int kLow, int lane) {
  int col = lane & 15;
  const h16* p = colbase + col * rowStride + kLow + ((lane & 16) ? 16 : 0);
  v16h bm;
  ((u32x4*)&bm)[0] = *(const u32x4*)p;
  ((u32x4*)&bm)[1] = *(const u32x4*)(p + 8);
  return bm;
}

// A operand (16 x 32) from LDS, row-contiguous: base[row*rowStride + k].
// lanes 0-15: row=lane, K 0-15; lanes 16-31: row=lane-16, K 16-31
__device__ __forceinline__ v16h load_A_lds(const h16* base, int rowStride, int lane) {
  int row = lane & 15;
  int kb  = (lane & 16) ? 16 : 0;
  const h16* p = base + row * rowStride + kb;
  v16h am;
  ((u32x4*)&am)[0] = *(const u32x4*)p;
  ((u32x4*)&am)[1] = *(const u32x4*)(p + 8);
  return am;
}

// A operand for Q loaded straight from global f32, scaled + converted to f16.
__device__ __forceinline__ v16h load_A_q(const float* rowp, int lane) {
  int kb = (lane & 16) ? 16 : 0;
  v16h am;
  h16* ap = (h16*)&am;
  const float* p = rowp + kb;
#pragma unroll
  for (int i = 0; i < 8; ++i) ap[i] = (h16)(p[i] * SCALE_);
#pragma unroll
  for (int i = 0; i < 8; ++i) ap[8 + i] = (h16)(p[8 + i] * SCALE_);
  return am;
}

// ---------------------------------------------------------------------------
// Kernel 2: one workgroup (256 thr = 8 wave32) per (bh, h).
// LDS: Ks[512][32] f16 (32KB) | Vt[32][512] f16 (32KB) | Pst 8x16x32 f16 (8KB)
//      rowmax2[2][64] | rowsum2[2][64] | Pred[4][2][8][32] f32 (8KB) -> 82944 B
// ---------------------------------------------------------------------------
__global__ __launch_bounds__(256) void sca_attn_kernel(const float* __restrict__ qptr,
                                                       const float* __restrict__ vptr,
                                                       const h16* __restrict__ ksum,
                                                       float* __restrict__ out) {
  extern __shared__ char smem[];
  h16*   Ks      = (h16*)smem;                 // [m][c]
  h16*   Vt      = Ks + 512 * HD_;             // [c][m] (transposed v band)
  h16*   Pst     = Vt + HD_ * 512;             // per-wave A-staging
  float* rowmax2 = (float*)(smem + 73728);     // [kh][row]
  float* rowsum2 = rowmax2 + 128;              // [kh][row]
  float* Pred    = rowsum2 + 128;              // [ri][nj][g][lane]

  const int tid  = threadIdx.x;
  const int lane = tid & 31;
  const int wv   = tid >> 5;
  const int ri   = wv >> 1;      // 16-row tile of the 64-row window
  const int kh   = wv & 1;       // 256-column K half

  const int bh   = blockIdx.x >> 3;
  const int h    = blockIdx.x & 7;
  const int b    = bh >> 3;
  const int head = bh & 7;

  // ---- Phase 0: stage k_sum and transposed v band into LDS (read once) ----
  {
    const u32x4* src = (const u32x4*)(ksum + (size_t)bh * 512 * HD_);
    u32x4* dst = (u32x4*)Ks;
#pragma unroll
    for (int i = 0; i < 8; ++i) dst[i * 256 + tid] = src[i * 256 + tid];
  }
  {
    const float* vbase = vptr + (size_t)b * L_ * DIM_ + head * HD_;
#pragma unroll
    for (int it = 0; it < 16; ++it) {
      int flat = it * 256 + tid;                  // (m, cgroup)
      int cg = flat & 7;
      int m  = flat >> 3;
      size_t sp = (size_t)((h * 8 + (m >> 6)) * 64 + (m & 63));
      f32x4 val = *(const f32x4*)(vbase + sp * DIM_ + cg * 4);
#pragma unroll
      for (int j = 0; j < 4; ++j) Vt[(cg * 4 + j) * 512 + m] = (h16)val[j];
    }
  }
  __syncthreads();

  const int rowlo = ri * 16 + ((lane & 16) ? 8 : 0);  // D-layout row base (+g)
  const size_t qb = (size_t)b * L_ * DIM_;
  const int n = ri * 16 + (lane & 15);                // this lane's window row
  const size_t qrow_off = (size_t)((h * 8 + (n >> 3)) * 64 + (n & 7)) * DIM_
                        + head * HD_;                 // + w*8*DIM_ per window

  for (int w = 0; w < 8; ++w) {
    // prefetch next window's Q rows while we compute this one
    if (w < 7)
      __builtin_prefetch(qptr + qb + qrow_off + (size_t)(w + 1) * 8 * DIM_, 0, 3);

    // ---- S = (Q*scale) @ k_sum^T : 16 tiles of 16x16, K=32 in one WMMA ----
    const float* qrow = qptr + qb + qrow_off + (size_t)w * 8 * DIM_;
    v16h aq = load_A_q(qrow, lane);

    v8f acc[16];
#pragma unroll
    for (int t = 0; t < 16; ++t) {
      v16h bk = load_B_lds(Ks + (kh * 256 + t * 16) * HD_, HD_, 0, lane);
      v8f z = {};
      acc[t] = wmma_f16(aq, bk, z);
    }

    // ---- row max (16 lanes per row half + partner-wave exchange) ----
#pragma unroll
    for (int g = 0; g < 8; ++g) {
      float mx = acc[0][g];
#pragma unroll
      for (int t = 1; t < 16; ++t) mx = fmaxf(mx, acc[t][g]);
#pragma unroll
      for (int off = 8; off >= 1; off >>= 1)
        mx = fmaxf(mx, __shfl_xor(mx, off, 32));
      if ((lane & 15) == 0) rowmax2[kh * 64 + rowlo + g] = mx;
    }
    __syncthreads();

    // ---- exp(S - rowmax), row sums ----
#pragma unroll
    for (int g = 0; g < 8; ++g) {
      int row = rowlo + g;
      float M = fmaxf(rowmax2[row], rowmax2[64 + row]);
      float s = 0.f;
#pragma unroll
      for (int t = 0; t < 16; ++t) {
        float e = __expf(acc[t][g] - M);
        acc[t][g] = e;
        s += e;
      }
#pragma unroll
      for (int off = 8; off >= 1; off >>= 1)
        s += __shfl_xor(s, off, 32);
      if ((lane & 15) == 0) rowsum2[kh * 64 + row] = s;
    }

    // ---- partial Out = P @ V over this wave's 256-wide K half ----
    v8f o0 = {}, o1 = {};
    h16* st = Pst + wv * (16 * 32);
#pragma unroll
    for (int t = 0; t < 8; ++t) {
      // stage two exp'd D-tiles (k-local 0..31) into per-wave A staging
#pragma unroll
      for (int tt = 0; tt < 2; ++tt) {
#pragma unroll
        for (int g = 0; g < 8; ++g) {
          int rl = g + ((lane & 16) ? 8 : 0);
          int kl = tt * 16 + (lane & 15);
          st[rl * 32 + kl] = (h16)acc[t * 2 + tt][g];
        }
      }
      v16h ap = load_A_lds(st, 32, lane);
      int kg = kh * 256 + t * 32;
      v16h b0 = load_B_lds(Vt,           512, kg, lane);
      v16h b1 = load_B_lds(Vt + 16 * 512, 512, kg, lane);
      o0 = wmma_f16(ap, b0, o0);
      o1 = wmma_f16(ap, b1, o1);
    }

    // ---- cross-K-half reduction, normalize, coalesced f32 store ----
    float* pr = Pred + ri * (2 * 8 * 32);
    if (kh == 1) {
#pragma unroll
      for (int g = 0; g < 8; ++g) {
        pr[g * 32 + lane]       = o0[g];
        pr[(8 + g) * 32 + lane] = o1[g];
      }
    }
    __syncthreads();
    if (kh == 0) {
#pragma unroll
      for (int g = 0; g < 8; ++g) {
        int row = rowlo + g;
        float inv = 1.0f / (rowsum2[row] + rowsum2[64 + row]);
        float v0 = (o0[g] + pr[g * 32 + lane]) * inv;
        float v1 = (o1[g] + pr[(8 + g) * 32 + lane]) * inv;
        float* op = out + (size_t)b * L_ * DIM_
                  + (size_t)((h * 8 + (row >> 3)) * 64 + w * 8 + (row & 7)) * DIM_
                  + head * HD_;
        op[lane & 15]        = v0;   // nj=0 -> channels head*32 + 0..15
        op[16 + (lane & 15)] = v1;   // nj=1 -> channels head*32 + 16..31
      }
    }
    __syncthreads();
  }
}

// ---------------------------------------------------------------------------
extern "C" void kernel_launch(void* const* d_in, const int* in_sizes, int n_in,
                              void* d_out, int out_size, void* d_ws, size_t ws_size,
                              hipStream_t stream) {
  (void)in_sizes; (void)n_in; (void)out_size; (void)ws_size;
  const float* qkv = (const float*)d_in[0];
  const float* q = qkv;
  const float* k = qkv + (size_t)1 * BATCH_ * L_ * DIM_;
  const float* v = qkv + (size_t)2 * BATCH_ * L_ * DIM_;
  float* out = (float*)d_out;
  h16* ws = (h16*)d_ws;  // 64 * 512 * 32 f16 = 2 MB

  sca_ksum_kernel<<<1024, 256, 0, stream>>>(k, ws);
  sca_attn_kernel<<<512, 256, 82944, stream>>>(q, v, ws, out);
}
